// MultiHeadAttention_89429809038131
// MI455X (gfx1250) — matrix operand order repaired and checked by reading, not compiled, not run on hardware
//
#include <hip/hip_runtime.h>

// Problem constants (B, S, D, H) = (4, 2048, 128, 8)
#define BB 4
#define SS 2048
#define DD 128
#define HH 8
#define HD (HH * DD)       // 1024
#define MROWS (BB * SS)    // 8192

typedef __attribute__((ext_vector_type(16))) __bf16 v16bf;
typedef __attribute__((ext_vector_type(8)))  __bf16 v8bf;
typedef __attribute__((ext_vector_type(8)))  float  v8f;

// ---------------------------------------------------------------------------
// Fragment loaders (wave32 WMMA; C/D per CDNA5 ISA 7.12.2, A/B use this file's contiguous k order)
// A 16x32 bf16: lane = half*16 + m ; elems 0..15 -> K = k0+half*16+e (contig)
// B 32x16 bf16: lane = half*16 + n ; elems 0..15 -> K = k0+half*16+e (contig)
// C/D 16x16 f32: lane = half*16 + n ; vgpr v -> row m = v + 8*half
// ---------------------------------------------------------------------------
static __device__ __forceinline__ v16bf fragA_bf16(const __bf16* p) {
  v8bf lo = *(const v8bf*)p;
  v8bf hi = *(const v8bf*)(p + 8);
  v16bf a;
#pragma unroll
  for (int e = 0; e < 8; ++e) { a[e] = lo[e]; a[8 + e] = hi[e]; }
  return a;
}

static __device__ __forceinline__ v16bf fragB_bf16(const __bf16* p) {
  return *(const v16bf*)p;   // 16 contiguous bf16 = 32B
}

static __device__ __forceinline__ v8f wmma_bf16(v16bf a, v16bf b, v8f c) {
  return __builtin_amdgcn_wmma_f32_16x16x32_bf16(false, a, false, b, (short)0, c,
                                                 false, false);
}

// Async global -> LDS copy of 16 bytes (gfx1250 ASYNCcnt path, GV addressing)
static __device__ __forceinline__ void async_b128(unsigned ldsoff, const void* g) {
  asm volatile("global_load_async_to_lds_b128 %0, %1, off"
               :: "v"(ldsoff), "v"(g) : "memory");
}

// ---------------------------------------------------------------------------
// Kernel 0: elementwise fp32 -> bf16 (amortizes conversions: weights/activations
// are each re-read many times by the GEMMs)
// ---------------------------------------------------------------------------
__global__ __launch_bounds__(256) void mha_cvt_bf16_kernel(
    const float* __restrict__ in, __bf16* __restrict__ outp, int n8) {
  int i = blockIdx.x * blockDim.x + threadIdx.x;
  if (i < n8) {
    v8f x = *(const v8f*)(in + (size_t)i * 8);
    v8bf y;
#pragma unroll
    for (int e = 0; e < 8; ++e) y[e] = (__bf16)x[e];
    *(v8bf*)(outp + (size_t)i * 8) = y;
  }
}

// ---------------------------------------------------------------------------
// Kernel 1: projection  Y = X @ W^T + bias, all-bf16 operands.
// One wave -> 16x64 output tile (A fragment reused across 4 WMMAs per k-chunk).
// mode 0: store [B,H,S,D];  mode 1: store [B,H,D,S] (V transposed for PV GEMM)
// ---------------------------------------------------------------------------
__global__ __launch_bounds__(256) void mha_qkv_proj_kernel(
    const __bf16* __restrict__ X, const __bf16* __restrict__ W,
    const float* __restrict__ bias, __bf16* __restrict__ Y, int mode) {
  const int lane = threadIdx.x & 31;
  const int half = lane >> 4;
  const int nl   = lane & 15;
  const int wave = (blockIdx.x * blockDim.x + threadIdx.x) >> 5;
  const int ngrp = HD / 64;                   // 16 groups of 4 n-tiles
  const int mtile = wave / ngrp;
  const int ng    = wave % ngrp;
  const int r = mtile * 16 + nl;              // A-fragment row (b*S+s)

  v8f acc[4] = {};
#pragma unroll
  for (int kc = 0; kc < 4; ++kc) {
    v16bf a = fragA_bf16(X + (size_t)r * DD + kc * 32 + half * 16);
#pragma unroll
    for (int j = 0; j < 4; ++j) {
      int c = ng * 64 + j * 16 + nl;          // W row = output column
      v16bf b = fragB_bf16(W + (size_t)c * DD + kc * 32 + half * 16);
      acc[j] = wmma_bf16(a, b, acc[j]);
    }
  }

#pragma unroll
  for (int j = 0; j < 4; ++j) {
    int c = ng * 64 + j * 16 + nl;
    float bv = bias[c];
    int h = c / DD, d = c % DD;
#pragma unroll
    for (int v = 0; v < 8; ++v) {
      int rr = mtile * 16 + v + 8 * half;
      int bi = rr / SS, s = rr % SS;
      float val = acc[j][v] + bv;
      size_t addr = (mode == 0)
          ? (((size_t)(bi * HH + h) * SS + s) * DD + d)     // [B,H,S,D]
          : (((size_t)(bi * HH + h) * DD + d) * SS + s);    // [B,H,D,S]
      Y[addr] = (__bf16)val;
    }
  }
}

// ---------------------------------------------------------------------------
// Kernel 2: flash attention. 8 waves/block share one (b,h); K/V kv-tiles (32
// rows) are double-buffer staged into LDS with async-to-LDS copies overlapped
// with compute. Each wave owns a 16-query tile: QK^T (8 WMMAs), online
// softmax, P restaged via LDS (C-layout -> A-fragment), O += P V (8 WMMAs).
// ---------------------------------------------------------------------------
__global__ __launch_bounds__(256) void mha_flash_attn_kernel(
    const __bf16* __restrict__ rQ,   // [B*H, S, D]
    const __bf16* __restrict__ rK,   // [B*H, S, D]
    const __bf16* __restrict__ rVT,  // [B*H, D, S]
    __bf16* __restrict__ oV) {       // [B, S, H*D]
  __shared__ __align__(16) __bf16 Kt[2][32 * DD];   // 2 x 8KB
  __shared__ __align__(16) __bf16 Vt[2][DD * 32];   // 2 x 8KB
  __shared__ __align__(16) __bf16 Pt[8][16 * 32];   // 8KB

  const int tid  = threadIdx.x;
  const int lane = tid & 31;
  const int half = lane >> 4;
  const int nl   = lane & 15;
  const int w    = tid >> 5;
  const int bh   = blockIdx.x >> 4;                 // 16 blocks per (b,h)
  const int s0   = (blockIdx.x & 15) * 128 + w * 16;
  const int b    = bh / HH, h = bh % HH;

  const __bf16* Qb = rQ  + (size_t)bh * SS * DD;
  const __bf16* Kb = rK  + (size_t)bh * SS * DD;
  const __bf16* Vb = rVT + (size_t)bh * DD * SS;

  // Preload Q fragments (16 rows x 128 head-dim = 4 A-fragments)
  v16bf qf[4];
#pragma unroll
  for (int kc = 0; kc < 4; ++kc)
    qf[kc] = fragA_bf16(Qb + (size_t)(s0 + nl) * DD + kc * 32 + half * 16);

  v8f o[8];
#pragma unroll
  for (int t = 0; t < 8; ++t) o[t] = (v8f){};
  float rowm[8], rowl[8];
#pragma unroll
  for (int v = 0; v < 8; ++v) { rowm[v] = -3.0e38f; rowl[v] = 0.0f; }

  const float sc = 0.08838834764831845f;            // 1/sqrt(128)
  const int nt = SS / 32;                           // 64 kv tiles

  // Prologue: stage tile 0 into buffer 0 (4 x 16B async copies per thread)
  {
#pragma unroll
    for (int i = 0; i < 2; ++i) {                   // K: 32x128 bf16 = 512 chunks
      int q = tid * 2 + i, row = q >> 4, c16 = q & 15;
      async_b128((unsigned)(size_t)(&Kt[0][0] + q * 8),
                 Kb + (size_t)row * DD + c16 * 8);
    }
#pragma unroll
    for (int i = 0; i < 2; ++i) {                   // V: 128x32 bf16 = 512 chunks
      int q = tid * 2 + i, d = q >> 2, c = q & 3;
      async_b128((unsigned)(size_t)(&Vt[0][0] + q * 8),
                 Vb + (size_t)d * SS + c * 8);
    }
  }

  for (int it = 0; it < nt; ++it) {
    const int cur = it & 1;
    asm volatile("s_wait_asynccnt 0" ::: "memory"); // this wave's tile copies done
    __syncthreads();                                // whole tile visible; old buf free

    if (it + 1 < nt) {                              // overlap next tile's staging
      const int kvn = (it + 1) * 32;
      __bf16* kd = &Kt[1 - cur][0];
      __bf16* vd = &Vt[1 - cur][0];
#pragma unroll
      for (int i = 0; i < 2; ++i) {
        int q = tid * 2 + i, row = q >> 4, c16 = q & 15;
        async_b128((unsigned)(size_t)(kd + q * 8),
                   Kb + (size_t)(kvn + row) * DD + c16 * 8);
      }
#pragma unroll
      for (int i = 0; i < 2; ++i) {
        int q = tid * 2 + i, d = q >> 2, c = q & 3;
        async_b128((unsigned)(size_t)(vd + q * 8),
                   Vb + (size_t)d * SS + kvn + c * 8);
      }
    }

    const __bf16* Kl = &Kt[cur][0];
    const __bf16* Vl = &Vt[cur][0];

    // Scores for 32 kv columns: two 16x16 C tiles (B-fragments from LDS)
    v8f sA = {}, sB = {};
#pragma unroll
    for (int kc = 0; kc < 4; ++kc) {
      v16bf b0 = fragB_bf16(Kl + (size_t)nl * DD + kc * 32 + half * 16);
      v16bf b1 = fragB_bf16(Kl + (size_t)(16 + nl) * DD + kc * 32 + half * 16);
      sA = wmma_bf16(qf[kc], b0, sA);
      sB = wmma_bf16(qf[kc], b1, sB);
    }

    // Online softmax per C-layout row (row m = v + 8*half; cols across lanes)
#pragma unroll
    for (int v = 0; v < 8; ++v) {
      float x0 = sA[v] * sc, x1 = sB[v] * sc;
      float mx = fmaxf(x0, x1);
#pragma unroll
      for (int off = 8; off; off >>= 1)
        mx = fmaxf(mx, __shfl_xor(mx, off, 16));
      float mnew = fmaxf(rowm[v], mx);
      float corr = __expf(rowm[v] - mnew);
      rowm[v] = mnew;
      float p0 = __expf(x0 - mnew);
      float p1 = __expf(x1 - mnew);
      float rs = p0 + p1;
#pragma unroll
      for (int off = 8; off; off >>= 1)
        rs += __shfl_xor(rs, off, 16);
      rowl[v] = rowl[v] * corr + rs;
#pragma unroll
      for (int t = 0; t < 8; ++t) o[t][v] *= corr;
      int m = v + 8 * half;
      Pt[w][m * 32 + nl]      = (__bf16)p0;
      Pt[w][m * 32 + 16 + nl] = (__bf16)p1;
    }

    asm volatile("s_wait_dscnt 0" ::: "memory");    // P store -> load ordering

    // Reload P as an A-fragment (16x32) and accumulate O += P V
    v16bf pf = fragA_bf16(&Pt[w][nl * 32 + half * 16]);
#pragma unroll
    for (int t = 0; t < 8; ++t) {
      v16bf bv = fragB_bf16(Vl + (size_t)(t * 16 + nl) * 32 + half * 16);
      o[t] = wmma_bf16(pf, bv, o[t]);
    }
    asm volatile("s_wait_dscnt 0" ::: "memory");    // drain before P overwrite
  }

  // Normalize and store merged heads: oV[b, s, h*D + d]
#pragma unroll
  for (int v = 0; v < 8; ++v) {
    float inv = 1.0f / rowl[v];
    int s = s0 + v + 8 * half;
#pragma unroll
    for (int t = 0; t < 8; ++t) {
      float val = o[t][v] * inv;
      oV[((size_t)b * SS + s) * HD + h * DD + t * 16 + nl] = (__bf16)val;
    }
  }
}

// ---------------------------------------------------------------------------
// Kernel 3: output projection  out = oV @ W_w^T + W_b (bf16 in, f32 out)
// One wave -> 16x128 output tile (A fragment reused across 8 WMMAs per k-chunk)
// ---------------------------------------------------------------------------
__global__ __launch_bounds__(256) void mha_out_proj_kernel(
    const __bf16* __restrict__ A, const __bf16* __restrict__ W,
    const float* __restrict__ bias, float* __restrict__ Y) {
  const int lane = threadIdx.x & 31;
  const int half = lane >> 4;
  const int nl   = lane & 15;
  const int mtile = (blockIdx.x * blockDim.x + threadIdx.x) >> 5;
  const int r = mtile * 16 + nl;

  v8f acc[8] = {};
#pragma unroll 4
  for (int kc = 0; kc < HD / 32; ++kc) {            // 32 k-chunks over K=1024
    v16bf a = fragA_bf16(A + (size_t)r * HD + kc * 32 + half * 16);
#pragma unroll
    for (int j = 0; j < 8; ++j) {
      int c = j * 16 + nl;
      v16bf b = fragB_bf16(W + (size_t)c * HD + kc * 32 + half * 16);
      acc[j] = wmma_bf16(a, b, acc[j]);
    }
  }
#pragma unroll
  for (int j = 0; j < 8; ++j) {
    int c = j * 16 + nl;
    float bv = bias[c];
#pragma unroll
    for (int v = 0; v < 8; ++v) {
      int rr = mtile * 16 + v + 8 * half;
      Y[(size_t)rr * DD + c] = acc[j][v] + bv;
    }
  }
}

// ---------------------------------------------------------------------------
// Launch
// ---------------------------------------------------------------------------
extern "C" void kernel_launch(void* const* d_in, const int* in_sizes, int n_in,
                              void* d_out, int out_size, void* d_ws, size_t ws_size,
                              hipStream_t stream) {
  const float* Q    = (const float*)d_in[0];
  const float* K    = (const float*)d_in[1];
  const float* V    = (const float*)d_in[2];
  const float* WQ_w = (const float*)d_in[3];
  const float* WQ_b = (const float*)d_in[4];
  const float* WK_w = (const float*)d_in[5];
  const float* WK_b = (const float*)d_in[6];
  const float* WV_w = (const float*)d_in[7];
  const float* WV_b = (const float*)d_in[8];
  const float* W_w  = (const float*)d_in[9];
  const float* W_b  = (const float*)d_in[10];
  float* out = (float*)d_out;

  const size_t BHSD  = (size_t)BB * HH * SS * DD;   // 8,388,608 elems
  const size_t NX    = (size_t)MROWS * DD;          // 1,048,576 elems
  const size_t NW    = (size_t)HD * DD;             //   131,072 elems
  char* ws = (char*)d_ws;
  size_t off = 0;
  __bf16* rQ  = (__bf16*)(ws + off); off += 2 * BHSD;
  __bf16* rK  = (__bf16*)(ws + off); off += 2 * BHSD;
  __bf16* rVT = (__bf16*)(ws + off); off += 2 * BHSD;
  __bf16* oV  = (__bf16*)(ws + off); off += 2 * BHSD;
  __bf16* Xq  = (__bf16*)(ws + off); off += 2 * NX;
  __bf16* Xk  = (__bf16*)(ws + off); off += 2 * NX;
  __bf16* Xv  = (__bf16*)(ws + off); off += 2 * NX;
  __bf16* Wq  = (__bf16*)(ws + off); off += 2 * NW;
  __bf16* Wk  = (__bf16*)(ws + off); off += 2 * NW;
  __bf16* Wv  = (__bf16*)(ws + off); off += 2 * NW;
  __bf16* Ww  = (__bf16*)(ws + off); off += 2 * NW;

  dim3 blk(256);

  // fp32 -> bf16 pre-conversion (amortized across all GEMM reuse)
  const int nx8 = (int)(NX / 8), nw8 = (int)(NW / 8);
  mha_cvt_bf16_kernel<<<(nx8 + 255) / 256, blk, 0, stream>>>(Q, Xq, nx8);
  mha_cvt_bf16_kernel<<<(nx8 + 255) / 256, blk, 0, stream>>>(K, Xk, nx8);
  mha_cvt_bf16_kernel<<<(nx8 + 255) / 256, blk, 0, stream>>>(V, Xv, nx8);
  mha_cvt_bf16_kernel<<<(nw8 + 255) / 256, blk, 0, stream>>>(WQ_w, Wq, nw8);
  mha_cvt_bf16_kernel<<<(nw8 + 255) / 256, blk, 0, stream>>>(WK_w, Wk, nw8);
  mha_cvt_bf16_kernel<<<(nw8 + 255) / 256, blk, 0, stream>>>(WV_w, Wv, nw8);
  mha_cvt_bf16_kernel<<<(nw8 + 255) / 256, blk, 0, stream>>>(W_w,  Ww, nw8);

  // QKV projections (16x64 tile per wave)
  const int proj_blocks = (MROWS / 16) * (HD / 64) / 8;   // 1024
  mha_qkv_proj_kernel<<<proj_blocks, blk, 0, stream>>>(Xq, Wq, WQ_b, rQ, 0);
  mha_qkv_proj_kernel<<<proj_blocks, blk, 0, stream>>>(Xk, Wk, WK_b, rK, 0);
  mha_qkv_proj_kernel<<<proj_blocks, blk, 0, stream>>>(Xv, Wv, WV_b, rVT, 1);

  // Flash attention (8 waves/block share a (b,h); async double-buffered K/V)
  const int attn_blocks = BB * HH * 16;                   // 512
  mha_flash_attn_kernel<<<attn_blocks, blk, 0, stream>>>(rQ, rK, rVT, oV);

  // Output projection (16x128 tile per wave)
  const int op_blocks = (MROWS / 16) / 8;                 // 64
  mha_out_proj_kernel<<<op_blocks, blk, 0, stream>>>(oV, Ww, W_b, out);
}
